// DRCGCN_74921409511626
// MI455X (gfx1250) — matrix operand order repaired and checked by reading, not compiled, not run on hardware
//
#include <hip/hip_runtime.h>
#include <hip/hip_bf16.h>

// ---------------------------------------------------------------------------
// DRC-GCN forward on MI455X (gfx1250, wave32, WMMA)
//
// Dense GEMMs: v_wmma_f32_16x16x32_bf16 with hi/lo bf16 split (3 WMMAs per
// K=32 step, ~fp32 accuracy). Weights are pre-swizzled ONCE into the exact
// per-lane WMMA B-fragment layout (hi/lo bf16) so B loads are contiguous
// b128 loads with zero conversion VALU. Each wave computes a full 16xN
// output strip so the A-fragment fp32->bf16 split happens once per K-step
// instead of once per column tile.
//
// SpMM: L2-resident gather + f32 atomic scatter (feature matrix = 100MB,
// fits in the 192MB L2).
// ---------------------------------------------------------------------------

#define NNODES 100000
#define NEDGES 3200000
#define NFEAT  512
#define NHID   256
#define NCLASS 64
#define NLAYER 8

typedef __attribute__((ext_vector_type(16))) __bf16 v16bf;
typedef __attribute__((ext_vector_type(8)))  float  v8f;

__device__ __forceinline__ v8f wmma_bf16(v16bf a, v16bf b, v8f c) {
    // (neg_a, A, neg_b, B, c_mod, C, reuse_a, reuse_b)
    return __builtin_amdgcn_wmma_f32_16x16x32_bf16(false, a, false, b,
                                                   (short)0, c, false, false);
}

// A fragment: 16x32 (MxK) block at src (row-major, leading dim lda), fp32 ->
// bf16 hi/lo split. ISA layout: lanes 0-15 M=lane, kbase=0; lanes 16-31
// M=lane-16, kbase=8. Element j: K = (j<8 ? kb+j : 16+kb+j-8).
__device__ __forceinline__ void load_a_frag(const float* __restrict__ src,
                                            int lda, int lane,
                                            v16bf& hi, v16bf& lo) {
    const int m  = lane & 15;
    const int kb = (lane >> 4) << 3;            // 0 or 8
    const float* p0 = src + (size_t)m * lda + kb;
    const float* p1 = p0 + 16;
#pragma unroll
    for (int j = 0; j < 8; ++j) {
        float x = p0[j];                        // contiguous 32B run -> b128
        __bf16 h = (__bf16)x;
        hi[j] = h;
        lo[j] = (__bf16)(x - (float)h);
        float y = p1[j];
        __bf16 h2 = (__bf16)y;
        hi[8 + j] = h2;
        lo[8 + j] = (__bf16)(y - (float)h2);
    }
}

// ---------------------------------------------------------------------------
// Weight pre-swizzle: f32 W[K x Ncols] -> bf16 hi/lo fragments laid out
// as F[tk][tn][lane][2][16] matching the 32x16 B-operand VGPR layout:
//   lanes 0-15: N=lane,    K=(j<8 ? j : 16+j-8)
//   lanes 16-31: N=lane-16, K=(j<8 ? 8+j : 24+j-8)
// ---------------------------------------------------------------------------
__global__ __launch_bounds__(256) void k_swz_w(const float* __restrict__ W,
                                               __bf16* __restrict__ F,
                                               int K, int Ncols) {
    int t = blockIdx.x * 256 + threadIdx.x;
    if (t >= K * Ncols) return;
    int k = t / Ncols;
    int n = t - k * Ncols;
    float x = W[t];
    __bf16 h = (__bf16)x;
    __bf16 l = (__bf16)(x - (float)h);
    int tk = k >> 5, kin = k & 31;
    int tn = n >> 4, nin = n & 15;
    int lane = (((kin >> 3) & 1) << 4) | nin;   // 0..31
    int j = (kin & 7) | ((kin >> 4) << 3);
    size_t base = ((((size_t)tk * (Ncols >> 4) + tn) * 32 + lane) * 2) * 16;
    F[base + j]      = h;
    F[base + 16 + j] = l;
}

// ---------------------------------------------------------------------------
// Core strip GEMM: one wave computes a 16 x (TN*16) output strip.
// A (f32, lda==KDIM) is split to bf16 hi/lo once per K-step; B comes from
// pre-swizzled fragments (pure b128 loads).
// ---------------------------------------------------------------------------
template <int KDIM, int TN>
__device__ __forceinline__ void gemm_strip(const float* __restrict__ A,
                                           const __bf16* __restrict__ Bfrag,
                                           int lane, v8f* cacc) {
    for (int k0 = 0; k0 < KDIM; k0 += 32) {
        v16bf ah, al;
        load_a_frag(A + k0, KDIM, lane, ah, al);
        const int tk = k0 >> 5;
        const __bf16* fk = Bfrag + (((size_t)tk * TN) * 32) * 32;
#pragma unroll
        for (int tn = 0; tn < TN; ++tn) {
            const __bf16* fp = fk + (((size_t)tn * 32 + lane) * 2) * 16;
            v16bf bh = *(const v16bf*)(fp);
            v16bf bl = *(const v16bf*)(fp + 16);
            cacc[tn] = wmma_bf16(ah, bh, cacc[tn]);
            cacc[tn] = wmma_bf16(ah, bl, cacc[tn]);
            cacc[tn] = wmma_bf16(al, bh, cacc[tn]);
        }
    }
}

// ---------------------------------------------------------------------------
// Kernel 1: H = X @ W_init + b_init ; write H to both Xc and acc.
// One wave per 16-row strip: 6250 waves, 8 waves/block.
// ---------------------------------------------------------------------------
__global__ __launch_bounds__(256) void k_init_gemm(
        const float* __restrict__ X, const __bf16* __restrict__ Wf,
        const float* __restrict__ bias, float* __restrict__ Xc,
        float* __restrict__ acc) {
    const int wave = threadIdx.x >> 5;
    const int lane = threadIdx.x & 31;
    const int tm = blockIdx.x * 8 + wave;
    if (tm >= NNODES / 16) return;

    const int n  = lane & 15;
    const int r0 = (lane >> 4) << 3;
    v8f c[16];
#pragma unroll
    for (int tn = 0; tn < 16; ++tn) {
        const float bv = bias[tn * 16 + n];
#pragma unroll
        for (int r = 0; r < 8; ++r) c[tn][r] = bv;
    }

    gemm_strip<NFEAT, 16>(X + (size_t)tm * 16 * NFEAT, Wf, lane, c);

    float* xp = Xc  + (size_t)tm * 16 * NHID;
    float* ap = acc + (size_t)tm * 16 * NHID;
#pragma unroll
    for (int tn = 0; tn < 16; ++tn) {
#pragma unroll
        for (int r = 0; r < 8; ++r) {
            const size_t idx = (size_t)(r0 + r) * NHID + tn * 16 + n;
            xp[idx] = c[tn][r];
            ap[idx] = c[tn][r];
        }
    }
}

// ---------------------------------------------------------------------------
// Kernel 2: zero AX (float4 stores)
// ---------------------------------------------------------------------------
__global__ __launch_bounds__(256) void k_zero(float4* __restrict__ p, long long n4) {
    long long i = (long long)blockIdx.x * blockDim.x + threadIdx.x;
    if (i < n4) p[i] = make_float4(0.f, 0.f, 0.f, 0.f);
}

// ---------------------------------------------------------------------------
// Kernel 3: SpMM  AX[row[e]] += w[e] * Xc[col[e]]
// One thread per (edge, 4-float feature chunk). Gather hits L2 (Xc = 100MB),
// scatter via global_atomic_add_f32 (AX = 100MB, L2 resident).
// ---------------------------------------------------------------------------
__global__ __launch_bounds__(256) void k_spmm(
        const int* __restrict__ erow, const int* __restrict__ ecol,
        const float* __restrict__ ew, const float* __restrict__ Xc,
        float* __restrict__ AX) {
    long long t = (long long)blockIdx.x * blockDim.x + threadIdx.x;
    int e = (int)(t >> 6);
    if (e >= NEDGES) return;
    int f = ((int)t & 63) << 2;
    int r = erow[e];
    int cidx = ecol[e];
    float w = ew[e];
    float4 x = *(const float4*)(Xc + (size_t)cidx * NHID + f);
    float* dst = AX + (size_t)r * NHID + f;
    atomicAdd(dst + 0, w * x.x);
    atomicAdd(dst + 1, w * x.y);
    atomicAdd(dst + 2, w * x.z);
    atomicAdd(dst + 3, w * x.w);
}

// ---------------------------------------------------------------------------
// Kernel 4: acc += (1-beta)*AX + beta*(AX @ W_l)
//   (algebraic fold of AX @ ((1-beta) I + beta W))
// ---------------------------------------------------------------------------
__global__ __launch_bounds__(256) void k_layer_gemm(
        const float* __restrict__ AX, const __bf16* __restrict__ Wf,
        float* __restrict__ acc, float beta) {
    const int wave = threadIdx.x >> 5;
    const int lane = threadIdx.x & 31;
    const int tm = blockIdx.x * 8 + wave;
    if (tm >= NNODES / 16) return;

    v8f c[16] = {};
    gemm_strip<NHID, 16>(AX + (size_t)tm * 16 * NHID, Wf, lane, c);

    const int n  = lane & 15;
    const int r0 = (lane >> 4) << 3;
    const float omb = 1.0f - beta;
    float* ap = acc + (size_t)tm * 16 * NHID;
    const float* axp = AX + (size_t)tm * 16 * NHID;
#pragma unroll
    for (int tn = 0; tn < 16; ++tn) {
#pragma unroll
        for (int r = 0; r < 8; ++r) {
            const size_t idx = (size_t)(r0 + r) * NHID + tn * 16 + n;
            ap[idx] += omb * axp[idx] + beta * c[tn][r];
        }
    }
}

// ---------------------------------------------------------------------------
// Kernel 5: Xc = gamma_l * (Xc - AX)   (elementwise, float4)
// ---------------------------------------------------------------------------
__global__ __launch_bounds__(256) void k_next_x(
        float4* __restrict__ Xc, const float4* __restrict__ AX,
        const float* __restrict__ gammas, int l, long long n4) {
    long long i = (long long)blockIdx.x * blockDim.x + threadIdx.x;
    if (i >= n4) return;
    const float g = gammas[l];
    float4 x = Xc[i];
    float4 a = AX[i];
    Xc[i] = make_float4(g * (x.x - a.x), g * (x.y - a.y),
                        g * (x.z - a.z), g * (x.w - a.w));
}

// ---------------------------------------------------------------------------
// Kernel 6: Z = acc @ W_sort + b_sort   [100000 x 64]
// ---------------------------------------------------------------------------
__global__ __launch_bounds__(256) void k_final_gemm(
        const float* __restrict__ acc, const __bf16* __restrict__ Wf,
        const float* __restrict__ bias, float* __restrict__ Z) {
    const int wave = threadIdx.x >> 5;
    const int lane = threadIdx.x & 31;
    const int tm = blockIdx.x * 8 + wave;
    if (tm >= NNODES / 16) return;

    const int n  = lane & 15;
    const int r0 = (lane >> 4) << 3;
    v8f c[4];
#pragma unroll
    for (int tn = 0; tn < 4; ++tn) {
        const float bv = bias[tn * 16 + n];
#pragma unroll
        for (int r = 0; r < 8; ++r) c[tn][r] = bv;
    }

    gemm_strip<NHID, 4>(acc + (size_t)tm * 16 * NHID, Wf, lane, c);

    float* zp = Z + (size_t)tm * 16 * NCLASS;
#pragma unroll
    for (int tn = 0; tn < 4; ++tn) {
#pragma unroll
        for (int r = 0; r < 8; ++r)
            zp[(size_t)(r0 + r) * NCLASS + tn * 16 + n] = c[tn][r];
    }
}

// ---------------------------------------------------------------------------
// Kernel 7: row-wise log_softmax over 64 classes. One wave per row,
// 2 columns per lane, shfl_xor reductions (wave32).
// ---------------------------------------------------------------------------
__global__ __launch_bounds__(256) void k_logsoftmax(
        const float* __restrict__ Z, float* __restrict__ out) {
    const int wave = threadIdx.x >> 5;
    const int lane = threadIdx.x & 31;
    const int row = blockIdx.x * 8 + wave;
    if (row >= NNODES) return;
    const float* z = Z + (size_t)row * NCLASS;
    float z0 = z[lane];
    float z1 = z[lane + 32];
    float m = fmaxf(z0, z1);
#pragma unroll
    for (int off = 16; off; off >>= 1)
        m = fmaxf(m, __shfl_xor(m, off, 32));
    float s = expf(z0 - m) + expf(z1 - m);
#pragma unroll
    for (int off = 16; off; off >>= 1)
        s += __shfl_xor(s, off, 32);
    const float lse = m + logf(s);
    float* o = out + (size_t)row * NCLASS;
    o[lane]      = z0 - lse;
    o[lane + 32] = z1 - lse;
}

// ---------------------------------------------------------------------------
// Host launcher
// ---------------------------------------------------------------------------
extern "C" void kernel_launch(void* const* d_in, const int* in_sizes, int n_in,
                              void* d_out, int out_size, void* d_ws, size_t ws_size,
                              hipStream_t stream) {
    const float* X      = (const float*)d_in[0];
    const int*   erow   = (const int*)  d_in[1];
    const int*   ecol   = (const int*)  d_in[2];
    const float* ew     = (const float*)d_in[3];
    const float* W_init = (const float*)d_in[4];
    const float* b_init = (const float*)d_in[5];
    const float* gammas = (const float*)d_in[6];
    const float* Ws     = (const float*)d_in[7];
    const float* W_sort = (const float*)d_in[8];
    const float* b_sort = (const float*)d_in[9];
    float* out = (float*)d_out;

    float* ws  = (float*)d_ws;
    const size_t feat_elems = (size_t)NNODES * NHID;   // 25.6M floats each
    float* Xc  = ws;
    float* acc = ws + feat_elems;
    float* AX  = ws + 2 * feat_elems;
    float* Z   = ws + 3 * feat_elems;                  // 100000 x 64

    // Pre-swizzled weight fragments (bf16 hi/lo), after Z (offset 32B-aligned)
    __bf16* Fb     = (__bf16*)(Z + (size_t)NNODES * NCLASS);
    __bf16* F_init = Fb;                                   // 512*256*2 bf16
    const size_t f_init_sz  = (size_t)NFEAT * NHID * 2;
    __bf16* F_lay  = F_init + f_init_sz;                   // 8 * 256*256*2
    const size_t f_lay_sz   = (size_t)NHID * NHID * 2;
    __bf16* F_sort = F_lay + NLAYER * f_lay_sz;            // 256*64*2

    const long long feat4 = (long long)NNODES * NHID / 4;     // 6.4M float4
    const int zero_blocks = (int)((feat4 + 255) / 256);       // 25000
    const long long spmm_threads = (long long)NEDGES * 64;
    const int spmm_blocks = (int)((spmm_threads + 255) / 256);
    const int strip_blocks = (NNODES / 16 + 7) / 8;           // 782

    // ---- one-time weight swizzle (cheap; runs inside the graph) ----
    k_swz_w<<<(NFEAT * NHID + 255) / 256, 256, 0, stream>>>(W_init, F_init,
                                                            NFEAT, NHID);
    for (int l = 0; l < NLAYER; ++l)
        k_swz_w<<<(NHID * NHID + 255) / 256, 256, 0, stream>>>(
            Ws + (size_t)l * NHID * NHID, F_lay + (size_t)l * f_lay_sz,
            NHID, NHID);
    k_swz_w<<<(NHID * NCLASS + 255) / 256, 256, 0, stream>>>(W_sort, F_sort,
                                                             NHID, NCLASS);

    // H = X @ W_init + b ; Xc = acc = H
    k_init_gemm<<<strip_blocks, 256, 0, stream>>>(X, F_init, b_init, Xc, acc);

    for (int l = 0; l < NLAYER; ++l) {
        const float beta = 0.5f / (float)(l + 1);
        k_zero<<<zero_blocks, 256, 0, stream>>>((float4*)AX, feat4);
        k_spmm<<<spmm_blocks, 256, 0, stream>>>(erow, ecol, ew, Xc, AX);
        k_layer_gemm<<<strip_blocks, 256, 0, stream>>>(
            AX, F_lay + (size_t)l * f_lay_sz, acc, beta);
        k_next_x<<<zero_blocks, 256, 0, stream>>>(
            (float4*)Xc, (const float4*)AX, gammas, l, feat4);
    }

    k_final_gemm<<<strip_blocks, 256, 0, stream>>>(acc, F_sort, b_sort, Z);
    k_logsoftmax<<<12500, 256, 0, stream>>>(Z, out);
}
